// stDecoder_13846974562433
// MI455X (gfx1250) — matrix operand order repaired and checked by reading, not compiled, not run on hardware
//
#include <hip/hip_runtime.h>
#include <cmath>

// ---- problem constants (from reference) ----
constexpr int Bc  = 32;    // batch
constexpr int Tt  = 20;    // frames
constexpr int Cc  = 512;   // channels
constexpr int Pp  = 196;   // pixels
constexpr int Hh  = 1024;  // hidden
constexpr int NCc = 101;   // classes

typedef __attribute__((ext_vector_type(16))) __bf16 v16bf;
typedef __attribute__((ext_vector_type(8)))  __bf16 v8bf;
typedef __attribute__((ext_vector_type(4)))  __bf16 v4bf;
typedef __attribute__((ext_vector_type(8)))  float  v8f;

// ---------------------------------------------------------------------------
// init: broadcast (1,H) initial states to (B,H), plus bf16 mirrors of h-states
// ---------------------------------------------------------------------------
__global__ void k_init(const float* __restrict__ h1, const float* __restrict__ c1,
                       const float* __restrict__ h2, const float* __restrict__ c2,
                       float* __restrict__ h1a, float* __restrict__ c1a,
                       float* __restrict__ h2a, float* __restrict__ c2a,
                       __bf16* __restrict__ h1b, __bf16* __restrict__ h2b) {
  int i = blockIdx.x * blockDim.x + threadIdx.x;
  if (i < Bc * Hh) {
    int h = i & (Hh - 1);
    float v1 = h1[h], v2 = h2[h];
    h1a[i] = v1; c1a[i] = c1[h]; h2a[i] = v2; c2a[i] = c2[h];
    h1b[i] = (__bf16)v1; h2b[i] = (__bf16)v2;
  }
}

// ---------------------------------------------------------------------------
// weight f32 -> bf16 conversion (once per launch; weights are read 20x)
// ---------------------------------------------------------------------------
__global__ __launch_bounds__(256) void k_cvt(const float* __restrict__ src,
                                             __bf16* __restrict__ dst, int n4) {
  int i = blockIdx.x * blockDim.x + threadIdx.x;
  if (i < n4) {
    float4 v = ((const float4*)src)[i];
    v4bf o = {(__bf16)v.x, (__bf16)v.y, (__bf16)v.z, (__bf16)v.w};
    ((v4bf*)dst)[i] = o;
  }
}

// ---------------------------------------------------------------------------
// sA2 precompute (hoisted out of the recurrence):
//   sA2[t,b,p] = sum_c videos[b,t,c,p] * W_sC21[c] + b_sC21
// ---------------------------------------------------------------------------
__global__ __launch_bounds__(256) void k_sA2(const float* __restrict__ videos,
                                             const float* __restrict__ wS,
                                             const float* __restrict__ bS,
                                             float* __restrict__ sA2) {
  __shared__ float w[Cc];
  int blk = blockIdx.x;
  int b = blk / Tt, t = blk % Tt;
  for (int c = threadIdx.x; c < Cc; c += blockDim.x) w[c] = wS[c];
  __syncthreads();
  int p = threadIdx.x;
  if (p < Pp) {
    const float* x = videos + (size_t)(b * Tt + t) * Cc * Pp + p;
    float acc = 0.f;
#pragma unroll 4
    for (int c = 0; c < Cc; ++c) acc += x[(size_t)c * Pp] * w[c];
    sA2[(size_t)(t * Bc + b) * Pp + p] = acc + bS[0];
  }
}

// ---------------------------------------------------------------------------
// attention step: sA1 = h2·W_h2p^T ; softmax -> alpha (direct to output);
// Y (bf16, feeds LSTM1 A-operand) ; beta_raw scalar per (t,b)
// ---------------------------------------------------------------------------
__global__ __launch_bounds__(256) void k_attn(
    int t, const float* __restrict__ videos, const float* __restrict__ h2cur,
    const float* __restrict__ W_h2p, const float* __restrict__ b_h2p,
    const float* __restrict__ sA2, const float* __restrict__ spatialBias,
    const float* __restrict__ W_h21, const float* __restrict__ b_h21,
    const float* __restrict__ W_tC21, const float* __restrict__ b_tC21,
    const float* __restrict__ temporalBias,
    float* __restrict__ alphas, __bf16* __restrict__ Yb,
    float* __restrict__ beta_raw) {
  __shared__ float h2s[Hh];
  __shared__ float alph[Pp];
  __shared__ float red[256];
  int b = blockIdx.x, tid = threadIdx.x;
  for (int k = tid; k < Hh; k += 256) h2s[k] = h2cur[b * Hh + k];
  __syncthreads();

  float s = -3.0e38f;
  int p = tid;
  if (p < Pp) {
    const float4* wr = (const float4*)(W_h2p + (size_t)p * Hh);
    float acc = 0.f;
#pragma unroll 4
    for (int k4 = 0; k4 < Hh / 4; ++k4) {
      float4 w = wr[k4];
      acc += w.x * h2s[4 * k4] + w.y * h2s[4 * k4 + 1] +
             w.z * h2s[4 * k4 + 2] + w.w * h2s[4 * k4 + 3];
    }
    s = acc + b_h2p[p] + sA2[(size_t)(t * Bc + b) * Pp + p] + spatialBias[p];
  }
  // softmax over P within the block
  red[tid] = s; __syncthreads();
  for (int off = 128; off; off >>= 1) {
    if (tid < off) red[tid] = fmaxf(red[tid], red[tid + off]);
    __syncthreads();
  }
  float mx = red[0]; __syncthreads();
  float e = (p < Pp) ? expf(s - mx) : 0.f;
  red[tid] = e; __syncthreads();
  for (int off = 128; off; off >>= 1) {
    if (tid < off) red[tid] += red[tid + off];
    __syncthreads();
  }
  float inv = 1.f / red[0]; __syncthreads();
  if (p < Pp) {
    float a = e * inv;
    alph[p] = a;
    alphas[(size_t)(t * Bc + b) * Pp + p] = a;  // direct to output
  }
  __syncthreads();

  // Y[b,c] = sum_p alpha[p] * x[b,t,c,p]; beta partials on the fly
  const float* xb = videos + (size_t)(b * Tt + t) * Cc * Pp;
  float bpart = 0.f;
  for (int c = tid; c < Cc; c += 256) {
    const float* row = xb + (size_t)c * Pp;
    __builtin_prefetch(row + 256 * Pp, 0, 0);  // next row for this thread
    float acc = 0.f;
#pragma unroll 4
    for (int q = 0; q < Pp; ++q) acc += alph[q] * row[q];
    Yb[b * Cc + c] = (__bf16)acc;
    bpart += acc * W_tC21[c];
  }
  for (int k = tid; k < Hh; k += 256) bpart += h2s[k] * W_h21[k];
  red[tid] = bpart; __syncthreads();
  for (int off = 128; off; off >>= 1) {
    if (tid < off) red[tid] += red[tid + off];
    __syncthreads();
  }
  if (tid == 0)
    beta_raw[t * Bc + b] = red[0] + b_h21[0] + b_tC21[0] + temporalBias[0];
}

// ---------------------------------------------------------------------------
// WMMA operand loaders. k order used here (A and B match):
// A 16x32 bf16: lane m=l&15; K=(l<16?0:16)+j contiguous (elts0-15)
// B 32x16 bf16: lane n=l&15; K=(l<16?0:16)+j contiguous; B[k][n]=W[n][k]
// shufflevector concat = pure register placement, loads feed WMMA directly.
// ---------------------------------------------------------------------------
__device__ __forceinline__ v16bf lda(const __bf16* __restrict__ p) {
  v8bf lo = *(const v8bf*)p;
  v8bf hi = *(const v8bf*)(p + 8);
  return __builtin_shufflevector(lo, hi, 0, 1, 2, 3, 4, 5, 6, 7, 8, 9, 10, 11,
                                 12, 13, 14, 15);
}
__device__ __forceinline__ v16bf ldb(const __bf16* __restrict__ p) {
  v8bf lo = *(const v8bf*)p;
  v8bf hi = *(const v8bf*)(p + 8);
  return __builtin_shufflevector(lo, hi, 0, 1, 2, 3, 4, 5, 6, 7, 8, 9, 10, 11,
                                 12, 13, 14, 15);
}
__device__ __forceinline__ v16bf ldb_f32(const float* __restrict__ p) {
  float4 w0 = *(const float4*)p;
  float4 w1 = *(const float4*)(p + 4);
  float4 w2 = *(const float4*)(p + 8);
  float4 w3 = *(const float4*)(p + 12);
  v16bf b;
  b[0] = (__bf16)w0.x;  b[1] = (__bf16)w0.y;  b[2] = (__bf16)w0.z;  b[3] = (__bf16)w0.w;
  b[4] = (__bf16)w1.x;  b[5] = (__bf16)w1.y;  b[6] = (__bf16)w1.z;  b[7] = (__bf16)w1.w;
  b[8] = (__bf16)w2.x;  b[9] = (__bf16)w2.y;  b[10] = (__bf16)w2.z; b[11] = (__bf16)w2.w;
  b[12] = (__bf16)w3.x; b[13] = (__bf16)w3.y; b[14] = (__bf16)w3.z; b[15] = (__bf16)w3.w;
  return b;
}
__device__ __forceinline__ v8f wmma_bf16(v16bf a, v16bf b, v8f c) {
  return __builtin_amdgcn_wmma_f32_16x16x32_bf16(false, a, false, b, (short)0,
                                                 c, false, false);
}

// one GEMM contribution: acc[g] += A(16xK) * W[g]^T tile
template <bool PREW>
__device__ __forceinline__ void gemm_part(v8f* acc,
    const __bf16* __restrict__ Xb,
    const __bf16* __restrict__ WB, const float* __restrict__ WF,
    int K, int arow, int ncol, int aoff, int boff) {
  const __bf16* pa = Xb + (size_t)arow * K + aoff;
  if constexpr (PREW) {
    const __bf16* pb0 = WB + (size_t)(0 * Hh + ncol) * K + boff;
    const __bf16* pb1 = WB + (size_t)(1 * Hh + ncol) * K + boff;
    const __bf16* pb2 = WB + (size_t)(2 * Hh + ncol) * K + boff;
    const __bf16* pb3 = WB + (size_t)(3 * Hh + ncol) * K + boff;
    for (int kk = 0; kk < K; kk += 32) {
      v16bf a  = lda(pa);
      v16bf b0 = ldb(pb0);
      v16bf b1 = ldb(pb1);
      v16bf b2 = ldb(pb2);
      v16bf b3 = ldb(pb3);
      acc[0] = wmma_bf16(a, b0, acc[0]);
      acc[1] = wmma_bf16(a, b1, acc[1]);
      acc[2] = wmma_bf16(a, b2, acc[2]);
      acc[3] = wmma_bf16(a, b3, acc[3]);
      pa += 32; pb0 += 32; pb1 += 32; pb2 += 32; pb3 += 32;
    }
  } else {
    const float* q0 = WF + (size_t)(0 * Hh + ncol) * K + boff;
    const float* q1 = WF + (size_t)(1 * Hh + ncol) * K + boff;
    const float* q2 = WF + (size_t)(2 * Hh + ncol) * K + boff;
    const float* q3 = WF + (size_t)(3 * Hh + ncol) * K + boff;
    for (int kk = 0; kk < K; kk += 32) {
      v16bf a  = lda(pa);
      v16bf b0 = ldb_f32(q0);
      v16bf b1 = ldb_f32(q1);
      v16bf b2 = ldb_f32(q2);
      v16bf b3 = ldb_f32(q3);
      acc[0] = wmma_bf16(a, b0, acc[0]);
      acc[1] = wmma_bf16(a, b1, acc[1]);
      acc[2] = wmma_bf16(a, b2, acc[2]);
      acc[3] = wmma_bf16(a, b3, acc[3]);
      pa += 32; q0 += 32; q1 += 32; q2 += 32; q3 += 32;
    }
  }
}

// ---------------------------------------------------------------------------
// fused LSTM cell: gates via WMMA (f32 acc), i/f/g/o accumulated per-wave so
// the cell nonlinearity runs in the epilogue with all 4 gates in registers.
// grid: 2 m-tiles * 64 n-tiles = 128 waves = 16 blocks * 256 threads.
// Uniform control flow -> EXEC all ones (WMMA requirement).
// ---------------------------------------------------------------------------
template <bool PREW>
__global__ __launch_bounds__(256) void k_lstm(
    const __bf16* __restrict__ Xb, int Kx,
    const __bf16* __restrict__ Hb, int Kh,
    const float* __restrict__ Cprev,
    const __bf16* __restrict__ WihB, const __bf16* __restrict__ WhhB,
    const float* __restrict__ WihF, const float* __restrict__ WhhF,
    const float* __restrict__ bih, const float* __restrict__ bhh,
    float* __restrict__ Hout, __bf16* __restrict__ HoutB,
    float* __restrict__ Cout) {
  int tid = threadIdx.x, lane = tid & 31;
  int wave = (blockIdx.x * blockDim.x + tid) >> 5;  // 0..127
  int mt = wave >> 6, nt = wave & 63;
  int m0 = mt * 16, n0 = nt * 16;
  int arow = m0 + (lane & 15);
  int ncol = n0 + (lane & 15);
  int aoff = (lane & 16) ? 16 : 0;
  int boff = (lane & 16) ? 16 : 0;

  v8f zero = {0.f, 0.f, 0.f, 0.f, 0.f, 0.f, 0.f, 0.f};
  v8f acc[4] = {zero, zero, zero, zero};

  gemm_part<PREW>(acc, Xb, WihB, WihF, Kx, arow, ncol, aoff, boff);
  gemm_part<PREW>(acc, Hb, WhhB, WhhF, Kh, arow, ncol, aoff, boff);

  // epilogue: C/D layout -> lane l, vgpr i: m = m0 + i + (l>=16 ? 8:0), h=ncol
  int h = ncol;
  float b0 = bih[h] + bhh[h];
  float b1 = bih[Hh + h] + bhh[Hh + h];
  float b2 = bih[2 * Hh + h] + bhh[2 * Hh + h];
  float b3 = bih[3 * Hh + h] + bhh[3 * Hh + h];
#pragma unroll
  for (int i = 0; i < 8; ++i) {
    int m = m0 + i + ((lane & 16) ? 8 : 0);
    float gi = acc[0][i] + b0;
    float gf = acc[1][i] + b1;
    float gg = acc[2][i] + b2;
    float go = acc[3][i] + b3;
    float si = 1.f / (1.f + expf(-gi));
    float sf = 1.f / (1.f + expf(-gf));
    float so = 1.f / (1.f + expf(-go));
    float cn = sf * Cprev[m * Hh + h] + si * tanhf(gg);
    float hn = so * tanhf(cn);
    Cout[m * Hh + h] = cn;
    Hout[m * Hh + h] = hn;
    HoutB[m * Hh + h] = (__bf16)hn;
  }
}

// ---------------------------------------------------------------------------
// classifier head: out[t,b,:] = h2n[b,:]·W_fc^T + b_fc  (tiny)
// ---------------------------------------------------------------------------
__global__ __launch_bounds__(128) void k_out(int t, const float* __restrict__ h2n,
                                             const float* __restrict__ W_fc,
                                             const float* __restrict__ b_fc,
                                             float* __restrict__ outbuf) {
  __shared__ float hs[Hh];
  int b = blockIdx.x, tid = threadIdx.x;
  for (int k = tid; k < Hh; k += 128) hs[k] = h2n[b * Hh + k];
  __syncthreads();
  if (tid < NCc) {
    const float4* wr = (const float4*)(W_fc + (size_t)tid * Hh);
    float acc = 0.f;
#pragma unroll 4
    for (int k4 = 0; k4 < Hh / 4; ++k4) {
      float4 w = wr[k4];
      acc += w.x * hs[4 * k4] + w.y * hs[4 * k4 + 1] +
             w.z * hs[4 * k4 + 2] + w.w * hs[4 * k4 + 3];
    }
    outbuf[(size_t)(t * Bc + b) * NCc + tid] = acc + b_fc[tid];
  }
}

// ---------------------------------------------------------------------------
// finalize: softmax betas over T -> betasT output; logits = beta-weighted sum
// ---------------------------------------------------------------------------
__global__ __launch_bounds__(128) void k_fin(const float* __restrict__ beta_raw,
                                             const float* __restrict__ outbuf,
                                             float* __restrict__ logits,
                                             float* __restrict__ betasT) {
  int b = blockIdx.x, tid = threadIdx.x;
  float mx = -3.0e38f;
  for (int t = 0; t < Tt; ++t) mx = fmaxf(mx, beta_raw[t * Bc + b]);
  float sm = 0.f;
  for (int t = 0; t < Tt; ++t) sm += expf(beta_raw[t * Bc + b] - mx);
  float inv = 1.f / sm;
  if (tid < Tt) betasT[b * Tt + tid] = expf(beta_raw[tid * Bc + b] - mx) * inv;
  for (int nc = tid; nc < NCc; nc += 128) {
    float acc = 0.f;
    for (int t = 0; t < Tt; ++t)
      acc += expf(beta_raw[t * Bc + b] - mx) * inv *
             outbuf[(size_t)(t * Bc + b) * NCc + nc];
    logits[b * NCc + nc] = acc;
  }
}

// ---------------------------------------------------------------------------
extern "C" void kernel_launch(void* const* d_in, const int* in_sizes, int n_in,
                              void* d_out, int out_size, void* d_ws,
                              size_t ws_size, hipStream_t stream) {
  (void)in_sizes; (void)n_in; (void)out_size;
  const float* videos       = (const float*)d_in[0];
  const float* h1_0         = (const float*)d_in[1];
  const float* c1_0         = (const float*)d_in[2];
  const float* h2_0         = (const float*)d_in[3];
  const float* c2_0         = (const float*)d_in[4];
  const float* spatialBias  = (const float*)d_in[5];
  const float* temporalBias = (const float*)d_in[6];
  const float* W_h2p        = (const float*)d_in[7];
  const float* b_h2p        = (const float*)d_in[8];
  const float* W_sC21       = (const float*)d_in[9];
  const float* b_sC21       = (const float*)d_in[10];
  const float* W_h21        = (const float*)d_in[11];
  const float* b_h21        = (const float*)d_in[12];
  const float* W_tC21       = (const float*)d_in[13];
  const float* b_tC21       = (const float*)d_in[14];
  const float* Wih1         = (const float*)d_in[15];
  const float* Whh1         = (const float*)d_in[16];
  const float* bih1         = (const float*)d_in[17];
  const float* bhh1         = (const float*)d_in[18];
  const float* Wih2         = (const float*)d_in[19];
  const float* Whh2         = (const float*)d_in[20];
  const float* bih2         = (const float*)d_in[21];
  const float* bhh2         = (const float*)d_in[22];
  const float* W_fc         = (const float*)d_in[23];
  const float* b_fc         = (const float*)d_in[24];

  float* out    = (float*)d_out;
  float* logits = out;                           // (B,NC)
  float* alphas = out + Bc * NCc;                // (T,B,P)
  float* betasT = alphas + (size_t)Tt * Bc * Pp; // (B,1,T)

  // ---- workspace layout ----
  float* ws = (float*)d_ws;
  float* h1A = ws; ws += Bc * Hh;
  float* h1B = ws; ws += Bc * Hh;
  float* c1A = ws; ws += Bc * Hh;
  float* c1B = ws; ws += Bc * Hh;
  float* h2A = ws; ws += Bc * Hh;
  float* h2B = ws; ws += Bc * Hh;
  float* c2A = ws; ws += Bc * Hh;
  float* c2B = ws; ws += Bc * Hh;
  float* sA2      = ws; ws += (size_t)Tt * Bc * Pp;
  float* outbuf   = ws; ws += (size_t)Tt * Bc * NCc;
  float* beta_raw = ws; ws += Tt * Bc;  // (640 floats; keeps next region aligned)

  __bf16* wb = (__bf16*)ws;
  __bf16* h1bA = wb; wb += Bc * Hh;
  __bf16* h1bB = wb; wb += Bc * Hh;
  __bf16* h2bA = wb; wb += Bc * Hh;
  __bf16* h2bB = wb; wb += Bc * Hh;
  __bf16* Yb   = wb; wb += Bc * Cc;

  const int nW1i = 4 * Hh * Cc;   // Wih1
  const int nWhh = 4 * Hh * Hh;   // Whh1 / Wih2 / Whh2
  __bf16* Wih1b = wb; wb += nW1i;
  __bf16* Whh1b = wb; wb += nWhh;
  __bf16* Wih2b = wb; wb += nWhh;
  __bf16* Whh2b = wb; wb += nWhh;

  const size_t need_full = (size_t)((char*)wb - (char*)d_ws);
  const bool prew = (ws_size >= need_full);

  k_init<<<(Bc * Hh + 255) / 256, 256, 0, stream>>>(
      h1_0, c1_0, h2_0, c2_0, h1A, c1A, h2A, c2A, h1bA, h2bA);
  k_sA2<<<Bc * Tt, 256, 0, stream>>>(videos, W_sC21, b_sC21, sA2);
  if (prew) {
    k_cvt<<<(nW1i / 4 + 255) / 256, 256, 0, stream>>>(Wih1, Wih1b, nW1i / 4);
    k_cvt<<<(nWhh / 4 + 255) / 256, 256, 0, stream>>>(Whh1, Whh1b, nWhh / 4);
    k_cvt<<<(nWhh / 4 + 255) / 256, 256, 0, stream>>>(Wih2, Wih2b, nWhh / 4);
    k_cvt<<<(nWhh / 4 + 255) / 256, 256, 0, stream>>>(Whh2, Whh2b, nWhh / 4);
  }

  float *h1c = h1A, *h1n = h1B, *c1c = c1A, *c1n = c1B;
  float *h2c = h2A, *h2n = h2B, *c2c = c2A, *c2n = c2B;
  __bf16 *h1bc = h1bA, *h1bn = h1bB, *h2bc = h2bA, *h2bn = h2bB;

  for (int t = 0; t < Tt; ++t) {
    k_attn<<<Bc, 256, 0, stream>>>(t, videos, h2c, W_h2p, b_h2p, sA2,
                                   spatialBias, W_h21, b_h21, W_tC21, b_tC21,
                                   temporalBias, alphas, Yb, beta_raw);
    if (prew) {
      k_lstm<true><<<16, 256, 0, stream>>>(Yb, Cc, h1bc, Hh, c1c, Wih1b, Whh1b,
                                           nullptr, nullptr, bih1, bhh1, h1n,
                                           h1bn, c1n);
      k_lstm<true><<<16, 256, 0, stream>>>(h1bn, Hh, h2bc, Hh, c2c, Wih2b,
                                           Whh2b, nullptr, nullptr, bih2, bhh2,
                                           h2n, h2bn, c2n);
    } else {
      k_lstm<false><<<16, 256, 0, stream>>>(Yb, Cc, h1bc, Hh, c1c, nullptr,
                                            nullptr, Wih1, Whh1, bih1, bhh1,
                                            h1n, h1bn, c1n);
      k_lstm<false><<<16, 256, 0, stream>>>(h1bn, Hh, h2bc, Hh, c2c, nullptr,
                                            nullptr, Wih2, Whh2, bih2, bhh2,
                                            h2n, h2bn, c2n);
    }
    k_out<<<Bc, 128, 0, stream>>>(t, h2n, W_fc, b_fc, outbuf);
    float* tp;
    tp = h1c; h1c = h1n; h1n = tp;
    tp = c1c; c1c = c1n; c1n = tp;
    tp = h2c; h2c = h2n; h2n = tp;
    tp = c2c; c2c = c2n; c2n = tp;
    __bf16* bp;
    bp = h1bc; h1bc = h1bn; h1bn = bp;
    bp = h2bc; h2bc = h2bn; h2bn = bp;
  }
  k_fin<<<Bc, 128, 0, stream>>>(beta_raw, outbuf, logits, betasT);
}
